// Top1Router_58720792871048
// MI455X (gfx1250) — matrix operand order repaired and checked by reading, not compiled, not run on hardware
//
#include <hip/hip_runtime.h>
#include <hip/hip_bf16.h>

// ---------------- problem constants (from reference) ----------------
#define HIDDEN       2048
#define NUM_EXPERTS  64
#define BATCH        4
#define SEQ          4096
#define TOKENS       (BATCH * SEQ)          // 16384
#define CAPACITY     128
#define CHUNK        256                    // tokens per capacity chunk (divides SEQ)
#define NCHUNKS      (TOKENS / CHUNK)       // 64
#define CHUNKS_PER_B (SEQ / CHUNK)          // 16
#define WAVES_PER_BLK 2                     // 64-thread blocks -> 512 GEMM blocks

typedef __bf16 v16bf __attribute__((ext_vector_type(16)));
typedef float  v8f   __attribute__((ext_vector_type(8)));

// ---------------- kernel 0: W (f32) -> bf16 once ----------------
__global__ __launch_bounds__(256) void cvt_w_kernel(const float* __restrict__ W,
                                                    __bf16* __restrict__ Wb, int n) {
  int i = blockIdx.x * 256 + threadIdx.x;
  if (i < n) Wb[i] = (__bf16)W[i];
}

// ---------------- kernel 1: router GEMM + softmax-max/argmax ----------------
// One wave computes a 16-token x 64-expert tile of logits with
// v_wmma_f32_16x16x32_bf16 (4 accumulators, K loop of 64 steps).
__global__ __launch_bounds__(WAVES_PER_BLK * 32) void router_gemm_kernel(
    const float* __restrict__ x, const __bf16* __restrict__ Wb,
    const float* __restrict__ bias, float* __restrict__ logits,
    float* __restrict__ probs, int* __restrict__ eidx) {
  __shared__ float tile[WAVES_PER_BLK][16][NUM_EXPERTS + 1];  // +1 pad: bank spread

  const int lane = threadIdx.x & 31;
  const int wave = threadIdx.x >> 5;
  const int row  = lane & 15;   // M (for A) / N (for B/C) within 16
  const int hi   = lane >> 4;   // lane-half selects K halves (contiguous k = 16*hi + s, same in A and B)
  const int token0 = (blockIdx.x * WAVES_PER_BLK + wave) * 16;

  const float* xrow = x + (size_t)(token0 + row) * HIDDEN;
  const int hi8 = hi * 8;

  v8f c0 = {}, c1 = {}, c2 = {}, c3 = {};

  for (int k0 = 0; k0 < HIDDEN; k0 += 32) {
    // A fragment: 16x32 bf16. Lane(row,hi): K = hi*16 + [0..15].
    float fa[16];
    *(float4*)(fa + 0)  = *(const float4*)(xrow + k0 + hi8 * 2);
    *(float4*)(fa + 4)  = *(const float4*)(xrow + k0 + hi8 * 2 + 4);
    *(float4*)(fa + 8)  = *(const float4*)(xrow + k0 + hi8 * 2 + 8);
    *(float4*)(fa + 12) = *(const float4*)(xrow + k0 + hi8 * 2 + 12);
    v16bf a;
#pragma unroll
    for (int i = 0; i < 16; ++i) a[i] = (__bf16)fa[i];

    // B fragments: 32x16 bf16 per expert-tile. Lane n holds W[n][k0 + hi*16 + 0..15].
    const __bf16* wp = Wb + (size_t)row * HIDDEN + k0 + hi * 16;  // 32B aligned
    v16bf b0 = *(const v16bf*)(wp);
    v16bf b1 = *(const v16bf*)(wp + 16 * HIDDEN);
    v16bf b2 = *(const v16bf*)(wp + 32 * HIDDEN);
    v16bf b3 = *(const v16bf*)(wp + 48 * HIDDEN);

    c0 = __builtin_amdgcn_wmma_f32_16x16x32_bf16(false, a, false, b0, (short)0, c0, false, false);
    c1 = __builtin_amdgcn_wmma_f32_16x16x32_bf16(false, a, false, b1, (short)0, c1, false, false);
    c2 = __builtin_amdgcn_wmma_f32_16x16x32_bf16(false, a, false, b2, (short)0, c2, false, false);
    c3 = __builtin_amdgcn_wmma_f32_16x16x32_bf16(false, a, false, b3, (short)0, c3, false, false);
  }

  // Epilogue: bias add, write logits, stage tile in LDS for per-token reduction.
  // C/D layout: VGPR r, lane -> (m = r + 8*hi, n = lane&15).
  const float bb0 = bias[row], bb1 = bias[16 + row], bb2 = bias[32 + row], bb3 = bias[48 + row];
#pragma unroll
  for (int r = 0; r < 8; ++r) {
    const int m = r + 8 * hi;
    float* lrow = &tile[wave][m][0];
    float* grow = logits + (size_t)(token0 + m) * NUM_EXPERTS;
    float v0 = c0[r] + bb0;  lrow[row]      = v0;  grow[row]      = v0;
    float v1 = c1[r] + bb1;  lrow[16 + row] = v1;  grow[16 + row] = v1;
    float v2 = c2[r] + bb2;  lrow[32 + row] = v2;  grow[32 + row] = v2;
    float v3 = c3[r] + bb3;  lrow[48 + row] = v3;  grow[48 + row] = v3;
  }
  __syncthreads();

  // Per-token: argmax (first-max, like jnp.argmax) and max softmax prob = 1/sum exp(l-max).
  if (lane < 16) {
    const int token = token0 + lane;
    float mx = -__builtin_inff();
    int arg = 0;
    for (int e = 0; e < NUM_EXPERTS; ++e) {
      float v = tile[wave][lane][e];
      if (v > mx) { mx = v; arg = e; }
    }
    float s = 0.f;
    for (int e = 0; e < NUM_EXPERTS; ++e) s += __expf(tile[wave][lane][e] - mx);
    probs[token] = 1.0f / s;
    eidx[token]  = arg;
  }
}

// ---------------- kernel 2a: per-chunk expert histogram ----------------
__global__ __launch_bounds__(CHUNK) void hist_kernel(const int* __restrict__ eidx,
                                                     int* __restrict__ counts) {
  __shared__ int h[NUM_EXPERTS];
  const int tid = threadIdx.x;
  if (tid < NUM_EXPERTS) h[tid] = 0;
  __syncthreads();
  atomicAdd(&h[eidx[blockIdx.x * CHUNK + tid]], 1);
  __syncthreads();
  if (tid < NUM_EXPERTS) counts[blockIdx.x * NUM_EXPERTS + tid] = h[tid];
}

// ---------------- kernel 2b: exclusive scan of chunk counts per (batch, expert) ----------------
__global__ __launch_bounds__(BATCH * NUM_EXPERTS) void scan_kernel(
    const int* __restrict__ counts, int* __restrict__ bases) {
  const int b = threadIdx.x >> 6;   // 0..3
  const int e = threadIdx.x & 63;   // 0..63
  int base = 0;
  for (int c = 0; c < CHUNKS_PER_B; ++c) {
    const int idx = (b * CHUNKS_PER_B + c) * NUM_EXPERTS + e;
    bases[idx] = base;
    base += counts[idx];
  }
}

// ---------------- kernel 2c: in-chunk rank, capacity mask, one-hot dispatch tile ----------------
__global__ __launch_bounds__(CHUNK) void dispatch_kernel(const int* __restrict__ eidx,
                                                         const int* __restrict__ bases,
                                                         float* __restrict__ out_ei) {
  __shared__ int eids[CHUNK];
  __shared__ int keeps[CHUNK];
  const int tid = threadIdx.x;
  const int e = eidx[blockIdx.x * CHUNK + tid];
  eids[tid] = e;
  __syncthreads();
  int rank = 1;  // inclusive cumsum position within the chunk
  for (int j = 0; j < tid; ++j) rank += (eids[j] == e);
  const int prio = bases[blockIdx.x * NUM_EXPERTS + e] + rank;
  keeps[tid] = (prio <= CAPACITY) ? 1 : 0;
  __syncthreads();
  // Coalesced write of the 256x64 one-hot tile.
  const size_t base_out = (size_t)blockIdx.x * CHUNK * NUM_EXPERTS;
#pragma unroll
  for (int i = 0; i < NUM_EXPERTS; ++i) {
    const int f  = i * CHUNK + tid;
    const int rl = f >> 6;
    const int ee = f & 63;
    out_ei[base_out + f] = (ee == eids[rl] && keeps[rl]) ? 1.0f : 0.0f;
  }
}

// ---------------- launch ----------------
extern "C" void kernel_launch(void* const* d_in, const int* in_sizes, int n_in,
                              void* d_out, int out_size, void* d_ws, size_t ws_size,
                              hipStream_t stream) {
  const float* x    = (const float*)d_in[0];
  const float* W    = (const float*)d_in[1];
  const float* bias = (const float*)d_in[2];

  float* out        = (float*)d_out;
  float* out_ei     = out;                                       // (B,S,E)  1,048,576
  float* out_probs  = out + (size_t)TOKENS * NUM_EXPERTS;        // (B,S,1)  16,384
  float* out_logits = out_probs + TOKENS;                        // (B,S,E)  1,048,576

  uint8_t* ws = (uint8_t*)d_ws;
  __bf16* Wb  = (__bf16*)ws;                                     // 262,144 B
  int* eidx   = (int*)(ws + 262144);                             //  65,536 B
  int* counts = (int*)(ws + 262144 + 65536);                     //  16,384 B
  int* bases  = (int*)(ws + 262144 + 65536 + 16384);             //  16,384 B

  cvt_w_kernel<<<(NUM_EXPERTS * HIDDEN + 255) / 256, 256, 0, stream>>>(
      W, Wb, NUM_EXPERTS * HIDDEN);
  router_gemm_kernel<<<TOKENS / (16 * WAVES_PER_BLK), WAVES_PER_BLK * 32, 0, stream>>>(
      x, Wb, bias, out_logits, out_probs, eidx);
  hist_kernel<<<NCHUNKS, CHUNK, 0, stream>>>(eidx, counts);
  scan_kernel<<<1, BATCH * NUM_EXPERTS, 0, stream>>>(counts, bases);
  dispatch_kernel<<<NCHUNKS, CHUNK, 0, stream>>>(eidx, bases, out_ei);
}
